// SE3TBackbone_48593259987060
// MI455X (gfx1250) — matrix operand order repaired and checked by reading, not compiled, not run on hardware
//
#include <hip/hip_runtime.h>
#include <math.h>

// SE(3)-Transformer backbone forward for MI455X (gfx1250, wave32, WMMA).
// B=16, N=64, FIN=16, C=32, D=4, L=4, H=8.

#define BQ   16
#define NN   64
#define FINd 16
#define CC   32
#define DD   4
#define LLAY 4
#define HH   8
#define BN   (BQ*NN)
#define EPSf 1e-6f
#define SUMM 16   // sum of (2l+1) for l=0..3; packed fiber width per channel

typedef __attribute__((ext_vector_type(16))) _Float16 v16h;
typedef __attribute__((ext_vector_type(8)))  float    v8f;

// ---------------------------------------------------------------------------
// Real spherical harmonics l=0..3, packed into 16 slots: offsets {0,1,4,9}
// ---------------------------------------------------------------------------
__device__ __forceinline__ void sph16(float x, float y, float z, float* Y) {
  const float c0 = 0.28209479177387814f;          // 0.5*sqrt(1/pi)
  const float c1 = 0.4886025119029199f;           // sqrt(3/(4pi))
  Y[0] = c0;
  Y[1] = c1 * y; Y[2] = c1 * z; Y[3] = c1 * x;
  const float a2 = 1.0925484305920792f;           // 0.5*sqrt(15/pi)
  const float b2 = 0.31539156525252005f;          // 0.25*sqrt(5/pi)
  const float e2 = 0.5462742152960396f;           // 0.25*sqrt(15/pi)
  Y[4] = a2 * x * y;
  Y[5] = a2 * y * z;
  Y[6] = b2 * (3.f * z * z - 1.f);
  Y[7] = a2 * x * z;
  Y[8] = e2 * (x * x - y * y);
  const float d0 = 0.5900435899266435f;           // 0.25*sqrt(35/(2pi))
  const float d1 = 2.890611442640554f;            // 0.5*sqrt(105/pi)
  const float d2 = 0.4570457994644658f;           // 0.25*sqrt(21/(2pi))
  const float d3 = 0.3731763325901154f;           // 0.25*sqrt(7/pi)
  const float d5 = 1.445305721320277f;            // 0.25*sqrt(105/pi)
  Y[9]  = d0 * y * (3.f * x * x - y * y);
  Y[10] = d1 * x * y * z;
  Y[11] = d2 * y * (5.f * z * z - 1.f);
  Y[12] = d3 * z * (5.f * z * z - 3.f);
  Y[13] = d2 * x * (5.f * z * z - 1.f);
  Y[14] = d5 * z * (x * x - y * y);
  Y[15] = d0 * x * (x * x - y * y);
}

// ---------------------------------------------------------------------------
// Pre-pack rw2 (L,C,512) f32 into WMMA B-fragment order, f16:
//   w2h[((li*512)+col)*32 + k] = rw2[li][k][col]
// Each column's 32 K-values are contiguous (64 B) so a lane's fragment half
// (16 halves = 32 B, 32 B-aligned) is one vector load.
// ---------------------------------------------------------------------------
__global__ __launch_bounds__(256) void k_pack(const float* __restrict__ rw2,
                                              _Float16* __restrict__ w2h) {
  int idx = blockIdx.x * 256 + threadIdx.x;   // LLAY*512*32 = 65536
  if (idx >= LLAY * 512 * 32) return;
  int k   = idx & 31;
  int col = (idx >> 5) & 511;
  int li  = idx >> 14;
  w2h[idx] = (_Float16)rw2[((size_t)li * 32 + k) * 512 + col];
}

// ---------------------------------------------------------------------------
// 64x64 = A[64x32]f16 @ B[32x64]f16 via v_wmma_f32_16x16x32_f16.
// A in LDS (row major, ld=32); Bh pre-packed fragment layout (see k_pack).
// Result + bias scattered to sr[64][64] per the documented C/D layout.
// All 8 wave32s participate uniformly (EXEC all-ones around WMMA).
// ---------------------------------------------------------------------------
__device__ __forceinline__ void gemm_wmma(const _Float16* A,
                                          const _Float16* Bh,
                                          const float* bias, float (*sr)[64],
                                          int tid) {
  int wv = tid >> 5, lane = tid & 31;
  int lr = lane & 15;
  bool lo = lane < 16;
  int kb = lo ? 0 : 16;
  for (int tt = wv; tt < 16; tt += 8) {
    int tm = tt >> 2, tn = tt & 3;
    // --- A fragment: 16-bit A 16x32 layout ---
    int row = tm * 16 + lr;
    int ka = lo ? 0 : 16;
    v16h a;
#pragma unroll
    for (int e = 0; e < 8; ++e) {
      a[e]     = A[row * CC + ka + e];        // K = ka..ka+7
      a[e + 8] = A[row * CC + 8 + ka + e];    // K = ka+8..ka+15
    }
    // --- B fragment: one 32-byte vector load from packed layout ---
    int col = tn * 16 + lr;
    v16h b = *(const v16h*)(Bh + col * 32 + kb);
    v8f c = {};
    c = __builtin_amdgcn_wmma_f32_16x16x32_f16(false, a, false, b,
                                               (short)0, c, false, false);
    // --- D scatter: VGPR j -> rows j (lanes<16) / j+8 (lanes>=16) ---
    float bi = bias[col];
    int rbase = tm * 16 + (lo ? 0 : 8);
#pragma unroll
    for (int j = 0; j < 8; ++j) sr[rbase + j][col] = c[j] + bi;
  }
}

// ---------------------------------------------------------------------------
// Per-(b,dst) edge prep: geometry, Y_l, edge feats, radial layer-1 (LN+ReLU)
// ---------------------------------------------------------------------------
__device__ void edge_prep(int b, int dst, int tid, const float* x,
                          const int* bond, const float* rw1, const float* rb1,
                          int li, float (*sY)[16], float (*srin)[5],
                          float (*spre)[CC], _Float16 (*shm)[CC]) {
  if (tid < NN) {
    int s = tid;
    float dx = x[(b * NN + s) * 3 + 0] - x[(b * NN + dst) * 3 + 0];
    float dy = x[(b * NN + s) * 3 + 1] - x[(b * NN + dst) * 3 + 1];
    float dz = x[(b * NN + s) * 3 + 2] - x[(b * NN + dst) * 3 + 2];
    float dist = sqrtf(dx * dx + dy * dy + dz * dz + EPSf);
    float inv = 1.f / dist;
    sph16(dx * inv, dy * inv, dz * inv, sY[s]);
    srin[s][0] = dist;
    int bt = bond[(b * NN + dst) * NN + s];
    bt = bt < 0 ? 0 : (bt > 4 ? 4 : bt);
#pragma unroll
    for (int j = 0; j < 4; ++j) srin[s][1 + j] = (bt == j + 1) ? 1.f : 0.f;
  }
  __syncthreads();
  const float* w1 = rw1 + li * 5 * CC;
  const float* b1 = rb1 + li * CC;
  for (int idx = tid; idx < NN * CC; idx += 256) {
    int s = idx >> 5, c = idx & 31;
    float acc = b1[c];
#pragma unroll
    for (int i = 0; i < 5; ++i) acc += srin[s][i] * w1[i * CC + c];
    spre[s][c] = acc;
  }
  __syncthreads();
  if (tid < NN) {
    int s = tid;
    float mu = 0.f;
    for (int c = 0; c < CC; ++c) mu += spre[s][c];
    mu *= (1.f / CC);
    float var = 0.f;
    for (int c = 0; c < CC; ++c) { float d = spre[s][c] - mu; var += d * d; }
    var *= (1.f / CC);
    float rs = rsqrtf(var + EPSf);
    for (int c = 0; c < CC; ++c) {
      float v = (spre[s][c] - mu) * rs;
      shm[s][c] = (_Float16)(v > 0.f ? v : 0.f);
    }
  }
  __syncthreads();
}

// ---------------------------------------------------------------------------
// K0: node embedding f0 = h @ Wemb; higher degrees zero.
// f packed per node as [c][mm], mm in 0..15 with degree offsets l*l.
// ---------------------------------------------------------------------------
__global__ __launch_bounds__(256) void k_embed(const float* __restrict__ h,
                                               const float* __restrict__ Wemb,
                                               float* __restrict__ f) {
  int idx = blockIdx.x * 256 + threadIdx.x;
  if (idx >= BN * CC) return;
  int node = idx >> 5, c = idx & 31;
  float acc = 0.f;
#pragma unroll
  for (int i = 0; i < FINd; ++i) acc += h[node * FINd + i] * Wemb[i * CC + c];
  float* o = f + (size_t)node * (CC * SUMM) + c * SUMM;
  o[0] = acc;
#pragma unroll
  for (int mm = 1; mm < SUMM; ++mm) o[mm] = 0.f;
}

// ---------------------------------------------------------------------------
// K1: per-node projections for layer li:
//   sck/scv[node][l*32+c], idk/idv/q/self[node][c*16+mm]
// ---------------------------------------------------------------------------
__global__ __launch_bounds__(256) void k_proj(
    const float* __restrict__ fIn, const float* __restrict__ Wk,
    const float* __restrict__ Wv, const float* __restrict__ Wq,
    const float* __restrict__ Wself, float* __restrict__ qbuf,
    float* __restrict__ idk, float* __restrict__ idv,
    float* __restrict__ sck, float* __restrict__ scv,
    float* __restrict__ selfp, int li) {
  __shared__ float sf[CC * SUMM];
  int node = blockIdx.x;
  int tid = threadIdx.x;
  for (int i = tid; i < CC * SUMM; i += 256)
    sf[i] = fIn[(size_t)node * (CC * SUMM) + i];
  __syncthreads();
  for (int t = tid; t < 2304; t += 256) {
    if (t < 256) {  // scalar (0->l) paths from f0
      int which = t >> 7;
      int l = (t >> 5) & 3;
      int c = t & 31;
      const float* W = (which == 0 ? Wk : Wv) +
                       ((((size_t)li * DD + l) * 2 + 0) * CC + c) * CC;
      float acc = 0.f;
#pragma unroll
      for (int i = 0; i < CC; ++i) acc += W[i] * sf[i * SUMM];
      (which == 0 ? sck : scv)[(size_t)node * (DD * CC) + l * CC + c] = acc;
    } else {
      int u = t - 256;
      int arr = u >> 9;          // 0:idk 1:idv 2:q 3:self
      int v = u & 511;
      int c = v >> 4, mm = v & 15;
      int l = mm >= 9 ? 3 : (mm >= 4 ? 2 : (mm >= 1 ? 1 : 0));
      const float* W;
      float* out;
      if (arr == 0) { W = Wk + ((((size_t)li * DD + l) * 2 + 1) * CC + c) * CC; out = idk; }
      else if (arr == 1) { W = Wv + ((((size_t)li * DD + l) * 2 + 1) * CC + c) * CC; out = idv; }
      else if (arr == 2) { W = Wq + (((size_t)li * DD + l) * CC + c) * CC; out = qbuf; }
      else { W = Wself + (((size_t)li * DD + l) * CC + c) * CC; out = selfp; }
      float acc = 0.f;
#pragma unroll
      for (int i = 0; i < CC; ++i) acc += W[i] * sf[i * SUMM + mm];
      out[(size_t)node * (CC * SUMM) + c * SUMM + mm] = acc;
    }
  }
}

// ---------------------------------------------------------------------------
// K2a: per-(b,dst) block: radial MLP (WMMA) -> keys -> logits -> softmax.
// Writes attn[bd*512 + s*8 + h].
// ---------------------------------------------------------------------------
__global__ __launch_bounds__(256) void k_logits(
    const float* __restrict__ x, const int* __restrict__ bond,
    const float* __restrict__ rw1, const float* __restrict__ rb1,
    const _Float16* __restrict__ w2h, const float* __restrict__ rb2,
    const float* __restrict__ qbuf, const float* __restrict__ idk,
    const float* __restrict__ sck, float* __restrict__ attn, int li) {
  __shared__ float sY[NN][16];
  __shared__ float srin[NN][5];
  __shared__ float spre[NN][CC];
  __shared__ _Float16 shm[NN][CC];
  __shared__ float sr[NN][64];
  __shared__ float sq[CC * SUMM];
  __shared__ float slog[NN][HH];

  int bd = blockIdx.x;
  int b = bd >> 6, dst = bd & 63;
  int tid = threadIdx.x;

  for (int i = tid; i < CC * SUMM; i += 256) sq[i] = qbuf[(size_t)bd * (CC * SUMM) + i];
  for (int i = tid; i < NN * HH; i += 256) (&slog[0][0])[i] = 0.f;

  edge_prep(b, dst, tid, x, bond, rw1, rb1, li, sY, srin, spre, shm);

  for (int l = 0; l < DD; ++l) {
    int moff = l * l, M = 2 * l + 1;
    const _Float16* Bh = w2h + ((size_t)li * 512 + l * 128) * 32;   // cols r0|r1
    const float* bias = rb2 + (size_t)li * 512 + l * 128;
    gemm_wmma(&shm[0][0], Bh, bias, sr, tid);
    __syncthreads();
    for (int p = tid; p < NN * HH; p += 256) {
      int s = p >> 3, hh = p & 7;
      const float* idkn = idk + (size_t)(b * NN + s) * (CC * SUMM);
      const float* sckn = sck + (size_t)(b * NN + s) * (DD * CC) + l * CC;
      float acc = 0.f;
#pragma unroll
      for (int c2 = 0; c2 < 4; ++c2) {
        int c = hh * 4 + c2;
        float r0 = sr[s][c], r1 = sr[s][32 + c];
        float qY = 0.f, qid = 0.f;
        for (int m = 0; m < M; ++m) {
          float qv = sq[c * SUMM + moff + m];
          qY += qv * sY[s][moff + m];
          qid += qv * idkn[c * SUMM + moff + m];
        }
        acc += r0 * sckn[c] * qY + r1 * qid;
      }
      slog[s][hh] += acc;
    }
    __syncthreads();
  }
  const float scale = 0.125f;  // 1/sqrt(hd * sumM) = 1/8
  for (int p = tid; p < NN * HH; p += 256) {
    int s = p >> 3, hh = p & 7;
    slog[s][hh] = (s == dst) ? -1e9f : slog[s][hh] * scale;
  }
  __syncthreads();
  if (tid < HH) {
    int hh = tid;
    float mx = -1e30f;
    for (int s = 0; s < NN; ++s) mx = fmaxf(mx, slog[s][hh]);
    float sum = 0.f;
    for (int s = 0; s < NN; ++s) sum += expf(slog[s][hh] - mx);
    float inv = 1.f / sum;
    for (int s = 0; s < NN; ++s)
      attn[(size_t)bd * (NN * HH) + s * HH + hh] = expf(slog[s][hh] - mx) * inv;
  }
}

// ---------------------------------------------------------------------------
// K2b: per-(b,dst) block: radial MLP (WMMA) -> values -> attn-weighted sum
//      -> self skip -> NormSE3 gate -> fOut.
// ---------------------------------------------------------------------------
__global__ __launch_bounds__(256) void k_output(
    const float* __restrict__ x, const int* __restrict__ bond,
    const float* __restrict__ rw1, const float* __restrict__ rb1,
    const _Float16* __restrict__ w2h, const float* __restrict__ rb2,
    const float* __restrict__ scv, const float* __restrict__ idv,
    const float* __restrict__ selfp, const float* __restrict__ gnorm,
    const float* __restrict__ bnorm, const float* __restrict__ attn,
    float* __restrict__ fOut, int li) {
  __shared__ float sY[NN][16];
  __shared__ float srin[NN][5];
  __shared__ float spre[NN][CC];
  __shared__ _Float16 shm[NN][CC];
  __shared__ float sr[NN][64];
  __shared__ float sattn[NN][HH];
  __shared__ float so[CC][7];

  int bd = blockIdx.x;
  int b = bd >> 6, dst = bd & 63;
  int tid = threadIdx.x;

  for (int i = tid; i < NN * HH; i += 256)
    (&sattn[0][0])[i] = attn[(size_t)bd * (NN * HH) + i];

  edge_prep(b, dst, tid, x, bond, rw1, rb1, li, sY, srin, spre, shm);

  for (int l = 0; l < DD; ++l) {
    int moff = l * l, M = 2 * l + 1;
    const _Float16* Bh = w2h + ((size_t)li * 512 + l * 128 + 64) * 32;  // r2|r3
    const float* bias = rb2 + (size_t)li * 512 + l * 128 + 64;
    gemm_wmma(&shm[0][0], Bh, bias, sr, tid);
    __syncthreads();
    int cnt = CC * M;
    if (tid < cnt) {
      int c = tid / M, m = tid % M;
      int mm = moff + m, hh = c >> 2;
      float acc = 0.f;
      for (int s = 0; s < NN; ++s) {
        int ns = b * NN + s;
        float v = sr[s][c] * scv[(size_t)ns * (DD * CC) + l * CC + c] * sY[s][mm] +
                  sr[s][32 + c] * idv[(size_t)ns * (CC * SUMM) + c * SUMM + mm];
        acc += sattn[s][hh] * v;
      }
      acc += selfp[(size_t)bd * (CC * SUMM) + c * SUMM + mm];
      so[c][m] = acc;
    }
    __syncthreads();
    if (tid < CC) {
      int c = tid;
      float s2 = 0.f;
      for (int m = 0; m < M; ++m) { float t = so[c][m]; s2 += t * t; }
      float nrm = sqrtf(s2 + EPSf);
      float g = gnorm[((size_t)li * DD + l) * CC + c];
      float bb = bnorm[((size_t)li * DD + l) * CC + c];
      float act = g * nrm + bb;
      act = act > 0.f ? act : 0.f;
      float ratio = act / nrm;
      for (int m = 0; m < M; ++m)
        fOut[(size_t)bd * (CC * SUMM) + c * SUMM + moff + m] = so[c][m] * ratio;
    }
    __syncthreads();
  }
}

// ---------------------------------------------------------------------------
// K3: invariant readout + final linear + mean pool over nodes.
// ---------------------------------------------------------------------------
__global__ __launch_bounds__(128) void k_readout(const float* __restrict__ f,
                                                 const float* __restrict__ Wfinal,
                                                 const float* __restrict__ bfinal,
                                                 float* __restrict__ out) {
  __shared__ float sinv[128];
  int b = blockIdx.x;
  int o = threadIdx.x;  // 128 output channels
  float acc = 0.f;
  for (int n = 0; n < NN; ++n) {
    const float* fn = f + (size_t)(b * NN + n) * (CC * SUMM);
    int l = o >> 5, c = o & 31;
    float v;
    if (l == 0) v = fn[c * SUMM];
    else {
      int moff = l * l, M = 2 * l + 1;
      float s2 = 0.f;
      for (int m = 0; m < M; ++m) { float t = fn[c * SUMM + moff + m]; s2 += t * t; }
      v = sqrtf(s2 + EPSf);
    }
    __syncthreads();
    sinv[o] = v;
    __syncthreads();
    float a = 0.f;
#pragma unroll 4
    for (int i = 0; i < 128; ++i) a += sinv[i] * Wfinal[i * 128 + o];
    acc += a;
  }
  out[b * 128 + o] = acc * (1.f / NN) + bfinal[o];
}

// ---------------------------------------------------------------------------
extern "C" void kernel_launch(void* const* d_in, const int* in_sizes, int n_in,
                              void* d_out, int out_size, void* d_ws,
                              size_t ws_size, hipStream_t stream) {
  (void)in_sizes; (void)n_in; (void)out_size; (void)ws_size;
  const float* h      = (const float*)d_in[0];
  const float* x      = (const float*)d_in[1];
  const int*   bond   = (const int*)d_in[2];
  const float* Wemb   = (const float*)d_in[3];
  const float* rw1    = (const float*)d_in[4];
  const float* rb1    = (const float*)d_in[5];
  const float* rw2    = (const float*)d_in[6];
  const float* rb2    = (const float*)d_in[7];
  const float* Wk     = (const float*)d_in[8];
  const float* Wv     = (const float*)d_in[9];
  const float* Wq     = (const float*)d_in[10];
  const float* Wself  = (const float*)d_in[11];
  const float* gnorm  = (const float*)d_in[12];
  const float* bnorm  = (const float*)d_in[13];
  const float* Wfinal = (const float*)d_in[14];
  const float* bfinal = (const float*)d_in[15];

  float* ws = (float*)d_ws;
  const size_t NF = (size_t)BN * (CC * SUMM);  // 524288 floats per fiber buffer
  float* fA    = ws;
  float* fB    = fA + NF;
  float* qb    = fB + NF;
  float* idk   = qb + NF;
  float* idv   = idk + NF;
  float* selfp = idv + NF;
  float* sck   = selfp + NF;
  float* scv   = sck + (size_t)BN * (DD * CC);
  float* attn  = scv + (size_t)BN * (DD * CC);            // B*N*N*H floats
  _Float16* w2h = (_Float16*)(attn + (size_t)BQ * NN * NN * HH);  // 64K halves

  k_pack<<<(LLAY * 512 * 32 + 255) / 256, 256, 0, stream>>>(rw2, w2h);
  k_embed<<<(BN * CC + 255) / 256, 256, 0, stream>>>(h, Wemb, fA);

  float* fin = fA;
  float* fout = fB;
  for (int li = 0; li < LLAY; ++li) {
    k_proj<<<BN, 256, 0, stream>>>(fin, Wk, Wv, Wq, Wself, qb, idk, idv, sck,
                                   scv, selfp, li);
    k_logits<<<BN, 256, 0, stream>>>(x, bond, rw1, rb1, w2h, rb2, qb, idk, sck,
                                     attn, li);
    k_output<<<BN, 256, 0, stream>>>(x, bond, rw1, rb1, w2h, rb2, scv, idv,
                                     selfp, gnorm, bnorm, attn, fout, li);
    float* t = fin; fin = fout; fout = t;
  }
  k_readout<<<BQ, 128, 0, stream>>>(fin, Wfinal, bfinal, (float*)d_out);
}
